// TransformerClsToken_72756745994672
// MI455X (gfx1250) — matrix operand order repaired and checked by reading, not compiled, not run on hardware
//
#include <hip/hip_runtime.h>
#include <hip/hip_bf16.h>

typedef _Float16 half_t;
typedef __attribute__((ext_vector_type(16))) _Float16 v16h;
typedef __attribute__((ext_vector_type(8)))  float    v8f;

#define B_    4
#define NCH   256
#define NPT   2048
#define DIMD  256
#define KNN   16
#define POSH  64
#define HID   1024
#define NP1   2049   // n + cls
#define EPS_  1e-5f

// ---------------------------------------------------------------------------
// WMMA fragment helpers (contiguous k order k = 16*g + i, wave32)
// ---------------------------------------------------------------------------
__device__ __forceinline__ v8f wmma_f16(v16h a, v16h b, v8f c) {
    return __builtin_amdgcn_wmma_f32_16x16x32_f16(false, a, false, b, (short)0, c, false, false);
}
// A tile 16x32 f16: row = lane&15, halves i -> K = g*16 + i (contiguous)
__device__ __forceinline__ v16h frag_a(const half_t* row32, int g) {
    v16h a; const half_t* p = row32 + g * 16;
#pragma unroll
    for (int i = 0; i < 16; ++i) a[i] = p[i];
    return a;
}
// B tile 32x16 f16: col = lane&15, halves i -> K = g*16 + i (contiguous col-major)
__device__ __forceinline__ v16h frag_b(const half_t* colk) {
    v16h b;
#pragma unroll
    for (int i = 0; i < 16; ++i) b[i] = colk[i];
    return b;
}

// ---------------------------------------------------------------------------
// KNN: per query point keep top-16 smallest distances (incl. self) in regs
// ---------------------------------------------------------------------------
__global__ __launch_bounds__(256) void knn_kernel(const float* __restrict__ pos,
                                                  int* __restrict__ idxout) {
    int gid = blockIdx.x * 256 + threadIdx.x;
    if (gid >= B_ * NPT) return;
    int b = gid >> 11, i = gid & (NPT - 1);
    const float* px = pos + (size_t)b * 3 * NPT;
    float xi = px[i], yi = px[NPT + i], zi = px[2 * NPT + i];
    float bd[KNN]; int bi[KNN];
#pragma unroll
    for (int t = 0; t < KNN; ++t) { bd[t] = 3.4e38f; bi[t] = 0; }
    for (int j = 0; j < NPT; ++j) {
        float dx = xi - px[j], dy = yi - px[NPT + j], dz = zi - px[2 * NPT + j];
        float d = dx * dx + dy * dy + dz * dz;
        if (d < bd[KNN - 1]) {
            bd[KNN - 1] = d; bi[KNN - 1] = j;
#pragma unroll
            for (int t = KNN - 1; t > 0; --t) {
                if (bd[t] < bd[t - 1]) {
                    float td = bd[t]; bd[t] = bd[t - 1]; bd[t - 1] = td;
                    int   ti = bi[t]; bi[t] = bi[t - 1]; bi[t - 1] = ti;
                }
            }
        }
    }
#pragma unroll
    for (int t = 0; t < KNN; ++t) idxout[(size_t)gid * KNN + t] = bi[t];
}

// ---------------------------------------------------------------------------
// Weight prep
// ---------------------------------------------------------------------------
__global__ void cvt_f16_kernel(const float* __restrict__ s, half_t* __restrict__ d, int n) {
    int i = blockIdx.x * 256 + threadIdx.x;
    if (i < n) d[i] = (half_t)s[i];
}
// Fold BN scale into W (fp16 out) and bias (fp32 out): y = (Wx+b-m)*g/sqrt(v+eps)+beta
__global__ void fold_bn_f16_kernel(const float* __restrict__ W, const float* __restrict__ bias,
                                   const float* __restrict__ g, const float* __restrict__ beta,
                                   const float* __restrict__ m, const float* __restrict__ v,
                                   half_t* __restrict__ Wout, float* __restrict__ bout,
                                   int O, int I) {
    int i = blockIdx.x * 256 + threadIdx.x;
    if (i < O * I) {
        int o = i / I;
        float s = g[o] * rsqrtf(v[o] + EPS_);
        Wout[i] = (half_t)(W[i] * s);
    }
    if (i < O) {
        float s = g[i] * rsqrtf(v[i] + EPS_);
        bout[i] = (bias[i] - m[i]) * s + beta[i];
    }
}
__global__ void fold_bn_f32_kernel(const float* __restrict__ W, const float* __restrict__ bias,
                                   const float* __restrict__ g, const float* __restrict__ beta,
                                   const float* __restrict__ m, const float* __restrict__ v,
                                   float* __restrict__ Wout, float* __restrict__ bout,
                                   int O, int I) {
    int i = blockIdx.x * 256 + threadIdx.x;
    if (i < O * I) {
        int o = i / I;
        float s = g[o] * rsqrtf(v[o] + EPS_);
        Wout[i] = W[i] * s;
    }
    if (i < O) {
        float s = g[i] * rsqrtf(v[i] + EPS_);
        bout[i] = (bias[i] - m[i]) * s + beta[i];
    }
}
__global__ void set_cls_kernel(const float* __restrict__ cls_tok, float* __restrict__ hx) {
    int b = blockIdx.x, c = threadIdx.x;
    hx[((size_t)b * DIMD + c) * NP1] = cls_tok[c];
}
// cls attention position: all K logits identical -> uniform softmax -> agg = v_cls
__global__ void cls_agg_kernel(const float* __restrict__ value, const float* __restrict__ pos_emb,
                               float* __restrict__ agg) {
    int b = blockIdx.x, c = threadIdx.x;
    agg[((size_t)b * DIMD + c) * NP1] = value[((size_t)b * DIMD + c) * NP1] + pos_emb[c];
}

// ---------------------------------------------------------------------------
// Generic WMMA GEMM: Y(b,O,ldY) = W(OxI,fp16) @ X(b,I,ldX,fp32) + bias
// block = 256 thr (8 waves), computes 128(O) x 128(M) tile; K chunked by 32.
// mode 0: Y[b][row][m+colOff] = v
// mode 1: m==0 -> out2[b*256+row]=v ; else Y[b][row][m-1] = v + ident[b][row][m-1]
// ---------------------------------------------------------------------------
__global__ __launch_bounds__(256) void gemm_wmma_kernel(
        const half_t* __restrict__ W, const float* __restrict__ bias,
        const float* __restrict__ X, int I, int M, int ldX,
        float* __restrict__ Y, int ldY, int colOff, int mode,
        const float* __restrict__ ident, float* __restrict__ out2) {
    __shared__ half_t As[128 * 32];
    __shared__ half_t Bs[128 * 32];   // col-major: Bs[m*32 + k]
    const int b  = blockIdx.z;
    const int o0 = blockIdx.y * 128;
    const int m0 = blockIdx.x * 128;
    const int tid = threadIdx.x, wave = tid >> 5, lane = tid & 31;
    const int nl = lane & 15, g = lane >> 4;
    const float* Xb = X + (size_t)b * I * ldX;

    v8f acc[8];
#pragma unroll
    for (int nt = 0; nt < 8; ++nt) { v8f z = {}; acc[nt] = z; }

    for (int k0 = 0; k0 < I; k0 += 32) {
        __syncthreads();
        for (int t = tid; t < 128 * 32; t += 256) {          // stage W chunk
            int r = t >> 5, c = t & 31;
            As[t] = W[(size_t)(o0 + r) * I + k0 + c];
        }
        for (int t = tid; t < 32 * 128; t += 256) {          // stage X chunk (fp32 -> fp16, transpose)
            int k = t >> 7, m = t & 127;
            int gm = m0 + m;
            float v = (gm < M) ? Xb[(size_t)(k0 + k) * ldX + gm] : 0.f;
            Bs[m * 32 + k] = (half_t)v;
        }
        __syncthreads();
        v16h af = frag_a(&As[(wave * 16 + nl) * 32], g);
#pragma unroll
        for (int nt = 0; nt < 8; ++nt) {
            v16h bf = frag_b(&Bs[(nt * 16 + nl) * 32 + g * 16]);
            acc[nt] = wmma_f16(af, bf, acc[nt]);
        }
    }
#pragma unroll
    for (int nt = 0; nt < 8; ++nt) {
#pragma unroll
        for (int r = 0; r < 8; ++r) {
            int row = o0 + wave * 16 + r + 8 * g;
            int m   = m0 + nt * 16 + nl;
            if (m < M) {
                float v = acc[nt][r] + bias[row];
                if (mode == 0) {
                    Y[((size_t)b * 256 + row) * ldY + m + colOff] = v;
                } else {
                    if (m == 0) out2[b * 256 + row] = v;
                    else {
                        size_t o1 = ((size_t)b * 256 + row) * (size_t)NPT + (m - 1);
                        Y[o1] = v + ident[o1];
                    }
                }
            }
        }
    }
}

// ---------------------------------------------------------------------------
// Fused attention: per (batch, tile of 16 points) workgroup.
//   pe MLP (WMMA) -> qk build -> per point: MLP1 (WMMA), MLP2 (WMMA),
//   softmax over K, aggregation with v_all. All intermediates in LDS.
// Dynamic LDS: qk 128K + pe 128K + hid/hpe 32K + logits 16K + idx 1K = 305 KB
// ---------------------------------------------------------------------------
#define ATTN_SMEM (256*256*2 /*qk*/ + 256*256*2 /*pe*/ + 1024*16*2 /*hid*/ + 256*16*4 /*logit*/ + 256*4 /*idx*/)

__global__ __launch_bounds__(256) void attn_fused_kernel(
        const float* __restrict__ query, const float* __restrict__ key,
        const float* __restrict__ value, const float* __restrict__ pos,
        const int* __restrict__ idx,
        const float* __restrict__ posW1f, const float* __restrict__ posB1f,
        const half_t* __restrict__ posW2h, const float* __restrict__ posB2,
        const half_t* __restrict__ aW1h, const float* __restrict__ aB1f,
        const half_t* __restrict__ aW2h, const float* __restrict__ aB2,
        float* __restrict__ agg) {
    extern __shared__ char smem[];
    half_t* s_qk    = (half_t*)smem;                 // [col*256 + c]   col-major
    half_t* s_pe    = s_qk + 256 * 256;              // [col*256 + c]
    half_t* s_hid   = s_pe + 256 * 256;              // hpe[col*64+h] then hid[n*1024+row]
    float*  s_logit = (float*)(s_hid + 1024 * 16);   // [row*16 + n]
    int*    s_idx   = (int*)(s_logit + 256 * 16);    // [256]

    const int blk  = blockIdx.x;
    const int b    = blk >> 7;           // 128 tiles / batch
    const int p0   = (blk & 127) * 16;
    const int tid  = threadIdx.x;
    const int wave = tid >> 5, lane = tid & 31;
    const int nl   = lane & 15, g = lane >> 4;

    // --- neighbor indices for the 16-point tile -----------------------------
    s_idx[tid] = idx[((size_t)b * NPT + p0 + (tid >> 4)) * KNN + (tid & 15)];
    __syncthreads();

    // --- hpe = relu(bn(posW1 @ pos_rel)) : pure VALU (I=3) ------------------
    {
        const int col = tid;
        const int p = p0 + (col >> 4), j = s_idx[col];
        const float* pb = pos + (size_t)b * 3 * NPT;
        float r0 = pb[p] - pb[j];
        float r1 = pb[NPT + p] - pb[NPT + j];
        float r2 = pb[2 * NPT + p] - pb[2 * NPT + j];
        half_t* hp = s_hid + col * 64;
#pragma unroll
        for (int h = 0; h < POSH; ++h) {
            float v = posB1f[h] + posW1f[h * 3] * r0 + posW1f[h * 3 + 1] * r1 + posW1f[h * 3 + 2] * r2;
            hp[h] = (half_t)fmaxf(v, 0.f);
        }
    }
    __syncthreads();

    // --- pe = posW2(256x64) @ hpe(64x256) + posB2 -> s_pe (fp16) ------------
    {
#pragma unroll 1
        for (int nt = 0; nt < 16; ++nt) {
            v8f c0 = {}, c1 = {};
#pragma unroll
            for (int kk = 0; kk < 2; ++kk) {
                v16h bf = frag_b(s_hid + (nt * 16 + nl) * 64 + kk * 32 + g * 16);
                v16h a0 = frag_a(posW2h + (size_t)(wave * 32 + nl) * 64 + kk * 32, g);
                v16h a1 = frag_a(posW2h + (size_t)(wave * 32 + 16 + nl) * 64 + kk * 32, g);
                c0 = wmma_f16(a0, bf, c0);
                c1 = wmma_f16(a1, bf, c1);
            }
#pragma unroll
            for (int r = 0; r < 8; ++r) {
                int row0 = wave * 32 + r + 8 * g, row1 = row0 + 16;
                int col = nt * 16 + nl;
                s_pe[col * 256 + row0] = (half_t)(c0[r] + posB2[row0]);
                s_pe[col * 256 + row1] = (half_t)(c1[r] + posB2[row1]);
            }
        }
    }
    __syncthreads();

    // --- qk = query - key_gathered + pe -> s_qk fp16 ------------------------
    {
        const int col = tid;
        const int p = p0 + (col >> 4), j = s_idx[col];
        const float* qp = query + (size_t)b * DIMD * NP1 + (p + 1);
        const float* kp = key   + (size_t)b * DIMD * NP1 + (j + 1);
        half_t* qc = s_qk + col * 256;
        const half_t* pc = s_pe + col * 256;
        for (int c = 0; c < DIMD; ++c) {
            float v = qp[(size_t)c * NP1] - kp[(size_t)c * NP1] + (float)pc[c];
            qc[c] = (half_t)v;
        }
    }
    __syncthreads();

    // --- per point: MLP1 -> MLP2 -> softmax(K) -> aggregate -----------------
    for (int p = 0; p < 16; ++p) {
        // MLP1: hid(1024x16) = relu(aW1 @ qk_cols + aB1); wave strip = 128 rows
        v8f acc[8];
#pragma unroll
        for (int mt = 0; mt < 8; ++mt) { v8f z = {}; acc[mt] = z; }
#pragma unroll 1
        for (int kk = 0; kk < 8; ++kk) {
            v16h bf = frag_b(s_qk + (p * 16 + nl) * 256 + kk * 32 + g * 16);
#pragma unroll
            for (int mt = 0; mt < 8; ++mt) {
                v16h af = frag_a(aW1h + (size_t)(wave * 128 + mt * 16 + nl) * 256 + kk * 32, g);
                acc[mt] = wmma_f16(af, bf, acc[mt]);
            }
        }
#pragma unroll
        for (int mt = 0; mt < 8; ++mt)
#pragma unroll
            for (int r = 0; r < 8; ++r) {
                int row = wave * 128 + mt * 16 + r + 8 * g;
                s_hid[nl * 1024 + row] = (half_t)fmaxf(acc[mt][r] + aB1f[row], 0.f);
            }
        __syncthreads();

        // MLP2: logits(256x16) = aW2(256x1024) @ hid + aB2; wave strip = 32 rows
        v8f c0 = {}, c1 = {};
#pragma unroll 1
        for (int kk = 0; kk < 32; ++kk) {
            v16h bf = frag_b(s_hid + nl * 1024 + kk * 32 + g * 16);
            v16h a0 = frag_a(aW2h + (size_t)(wave * 32 + nl) * 1024 + kk * 32, g);
            v16h a1 = frag_a(aW2h + (size_t)(wave * 32 + 16 + nl) * 1024 + kk * 32, g);
            c0 = wmma_f16(a0, bf, c0);
            c1 = wmma_f16(a1, bf, c1);
        }
#pragma unroll
        for (int r = 0; r < 8; ++r) {
            int row0 = wave * 32 + r + 8 * g, row1 = row0 + 16;
            s_logit[row0 * 16 + nl] = c0[r] + aB2[row0];
            s_logit[row1 * 16 + nl] = c1[r] + aB2[row1];
        }
        __syncthreads();

        // softmax over K + weighted sum with v_all = value + pe
        {
            const int c = tid;
            float l[KNN], mx = -3.4e38f;
#pragma unroll
            for (int k = 0; k < KNN; ++k) { l[k] = s_logit[c * 16 + k]; mx = fmaxf(mx, l[k]); }
            float s = 0.f;
#pragma unroll
            for (int k = 0; k < KNN; ++k) { l[k] = __expf(l[k] - mx); s += l[k]; }
            float vv = value[((size_t)b * DIMD + c) * NP1 + (p0 + p + 1)];
            float a = 0.f;
#pragma unroll
            for (int k = 0; k < KNN; ++k) a += l[k] * (vv + (float)s_pe[(p * 16 + k) * 256 + c]);
            agg[((size_t)b * DIMD + c) * NP1 + (p0 + p + 1)] = a / s;
        }
        __syncthreads();
    }
}

// ---------------------------------------------------------------------------
// Host driver
// ---------------------------------------------------------------------------
extern "C" void kernel_launch(void* const* d_in, const int* in_sizes, int n_in,
                              void* d_out, int out_size, void* d_ws, size_t ws_size,
                              hipStream_t stream) {
    const float* x       = (const float*)d_in[0];
    const float* pos     = (const float*)d_in[1];
    const float* W_start = (const float*)d_in[2];
    const float* b_start = (const float*)d_in[3];
    const float* W_key   = (const float*)d_in[4];
    const float* b_key   = (const float*)d_in[5];
    const float* W_query = (const float*)d_in[6];
    const float* b_query = (const float*)d_in[7];
    const float* W_value = (const float*)d_in[8];
    const float* b_value = (const float*)d_in[9];
    const float* pos_W1  = (const float*)d_in[10];
    const float* pos_b1  = (const float*)d_in[11];
    const float* pos_g   = (const float*)d_in[12];
    const float* pos_beta= (const float*)d_in[13];
    const float* pos_m   = (const float*)d_in[14];
    const float* pos_v   = (const float*)d_in[15];
    const float* pos_W2  = (const float*)d_in[16];
    const float* pos_b2  = (const float*)d_in[17];
    const float* attn_W1 = (const float*)d_in[18];
    const float* attn_b1 = (const float*)d_in[19];
    const float* attn_g  = (const float*)d_in[20];
    const float* attn_beta=(const float*)d_in[21];
    const float* attn_m  = (const float*)d_in[22];
    const float* attn_v  = (const float*)d_in[23];
    const float* attn_W2 = (const float*)d_in[24];
    const float* attn_b2 = (const float*)d_in[25];
    const float* W_end   = (const float*)d_in[26];
    const float* b_end   = (const float*)d_in[27];
    const float* pos_emb = (const float*)d_in[28];
    const float* cls_tok = (const float*)d_in[29];

    char* ws = (char*)d_ws;
    size_t off = 0;
    auto alloc = [&](size_t bytes) -> void* {
        void* p = ws + off;
        off = (off + bytes + 255) & ~(size_t)255;
        return p;
    };

    int*    idx    = (int*)   alloc((size_t)B_ * NPT * KNN * 4);
    float*  hx     = (float*) alloc((size_t)B_ * DIMD * NP1 * 4);
    float*  keyb   = (float*) alloc((size_t)B_ * DIMD * NP1 * 4);
    float*  queryb = (float*) alloc((size_t)B_ * DIMD * NP1 * 4);
    float*  valueb = (float*) alloc((size_t)B_ * DIMD * NP1 * 4);
    float*  aggb   = (float*) alloc((size_t)B_ * DIMD * NP1 * 4);
    half_t* Ws16   = (half_t*)alloc((size_t)DIMD * NCH * 2);
    half_t* Wk16   = (half_t*)alloc((size_t)DIMD * DIMD * 2);
    half_t* Wq16   = (half_t*)alloc((size_t)DIMD * DIMD * 2);
    half_t* Wv16   = (half_t*)alloc((size_t)DIMD * DIMD * 2);
    half_t* We16   = (half_t*)alloc((size_t)NCH * DIMD * 2);
    half_t* pW2h   = (half_t*)alloc((size_t)DIMD * POSH * 2);
    half_t* aW1h   = (half_t*)alloc((size_t)HID * DIMD * 2);
    half_t* aW2h   = (half_t*)alloc((size_t)DIMD * HID * 2);
    float*  aB1f   = (float*) alloc((size_t)HID * 4);
    float*  pW1f   = (float*) alloc((size_t)POSH * 3 * 4);
    float*  pB1f   = (float*) alloc((size_t)POSH * 4);
    (void)n_in; (void)in_sizes; (void)out_size; (void)ws_size;

    float* out1 = (float*)d_out;                       // (B, NCH, NPT)
    float* out2 = (float*)d_out + (size_t)B_ * NCH * NPT; // (B, NCH, 1)

    // 1) KNN
    knn_kernel<<<(B_ * NPT + 255) / 256, 256, 0, stream>>>(pos, idx);

    // 2) Weight prep (fp16 conversions + BN folds)
    cvt_f16_kernel<<<(DIMD * NCH + 255) / 256, 256, 0, stream>>>(W_start, Ws16, DIMD * NCH);
    cvt_f16_kernel<<<(DIMD * DIMD + 255) / 256, 256, 0, stream>>>(W_key, Wk16, DIMD * DIMD);
    cvt_f16_kernel<<<(DIMD * DIMD + 255) / 256, 256, 0, stream>>>(W_query, Wq16, DIMD * DIMD);
    cvt_f16_kernel<<<(DIMD * DIMD + 255) / 256, 256, 0, stream>>>(W_value, Wv16, DIMD * DIMD);
    cvt_f16_kernel<<<(NCH * DIMD + 255) / 256, 256, 0, stream>>>(W_end, We16, NCH * DIMD);
    cvt_f16_kernel<<<(DIMD * POSH + 255) / 256, 256, 0, stream>>>(pos_W2, pW2h, DIMD * POSH);
    cvt_f16_kernel<<<(DIMD * HID + 255) / 256, 256, 0, stream>>>(attn_W2, aW2h, DIMD * HID);
    fold_bn_f16_kernel<<<(HID * DIMD + 255) / 256, 256, 0, stream>>>(
        attn_W1, attn_b1, attn_g, attn_beta, attn_m, attn_v, aW1h, aB1f, HID, DIMD);
    fold_bn_f32_kernel<<<1, 256, 0, stream>>>(
        pos_W1, pos_b1, pos_g, pos_beta, pos_m, pos_v, pW1f, pB1f, POSH, 3);

    // 3) hx = [cls_tok | W_start @ x + b_start]
    set_cls_kernel<<<B_, 256, 0, stream>>>(cls_tok, hx);
    {
        dim3 grid(NPT / 128, DIMD / 128, B_);
        gemm_wmma_kernel<<<grid, 256, 0, stream>>>(Ws16, b_start, x, NCH, NPT, NPT,
                                                   hx, NP1, 1, 0, nullptr, nullptr);
    }

    // 4) key / query / value = W @ hx + b   (M = 2049)
    {
        dim3 grid((NP1 + 127) / 128, DIMD / 128, B_);
        gemm_wmma_kernel<<<grid, 256, 0, stream>>>(Wk16, b_key, hx, DIMD, NP1, NP1,
                                                   keyb, NP1, 0, 0, nullptr, nullptr);
        gemm_wmma_kernel<<<grid, 256, 0, stream>>>(Wq16, b_query, hx, DIMD, NP1, NP1,
                                                   queryb, NP1, 0, 0, nullptr, nullptr);
        gemm_wmma_kernel<<<grid, 256, 0, stream>>>(Wv16, b_value, hx, DIMD, NP1, NP1,
                                                   valueb, NP1, 0, 0, nullptr, nullptr);
    }

    // 5) cls position shortcut: uniform softmax -> agg[:, 0] = v_cls
    cls_agg_kernel<<<B_, 256, 0, stream>>>(valueb, pos_emb, aggb);

    // 6) fused attention over 16-point tiles
    (void)hipFuncSetAttribute((const void*)attn_fused_kernel,
                              hipFuncAttributeMaxDynamicSharedMemorySize, ATTN_SMEM);
    attn_fused_kernel<<<B_ * (NPT / 16), 256, ATTN_SMEM, stream>>>(
        queryb, keyb, valueb, pos, idx, pW1f, pB1f, pW2h, pos_b2,
        aW1h, aB1f, aW2h, attn_b2, aggb);

    // 7) y = W_end @ agg + b_end ; out1 = y[:,1:] + x ; out2 = y[:,0]
    {
        dim3 grid((NP1 + 127) / 128, NCH / 128, B_);
        gemm_wmma_kernel<<<grid, 256, 0, stream>>>(We16, b_end, aggb, DIMD, NP1, NP1,
                                                   out1, NPT, 0, 1, x, out2);
    }
}
